// pointer_network_76622216561042
// MI455X (gfx1250) — matrix operand order repaired and checked by reading, not compiled, not run on hardware
//
#include <hip/hip_runtime.h>
#include <stdint.h>

// ---------------------------------------------------------------------------
// Problem constants (from the reference)
// ---------------------------------------------------------------------------
static constexpr int B_   = 512;
static constexpr int SEQ_ = 255;
static constexpr int D_   = 128;
static constexpr int H_   = 256;
static constexpr int WD_  = 256;
static constexpr int T_   = 32;
static constexpr int S_   = 256;   // SEQ + 1
static constexpr float NEG_INF_ = -1e10f;
static constexpr float EPS_ = 1e-9f;

// ---------------------------------------------------------------------------
// WMMA / TDM types (CDNA5 / gfx1250, wave32)
// ---------------------------------------------------------------------------
typedef __attribute__((ext_vector_type(16))) __bf16 v16bf;
typedef __attribute__((ext_vector_type(8)))  float  v8f;
typedef __attribute__((ext_vector_type(4)))  unsigned v4u;
typedef __attribute__((ext_vector_type(8)))  int    v8i;
typedef __attribute__((ext_vector_type(4)))  int    v4i;

union AFragU { v16bf v; unsigned u[8]; };
union BFragU { v16bf v; unsigned u[8]; };

__device__ __forceinline__ unsigned short f2bf(float f) {
  union { float f; unsigned u; } x; x.f = f;
  unsigned r = x.u + 0x7FFFu + ((x.u >> 16) & 1u);   // round-to-nearest-even
  return (unsigned short)(r >> 16);
}

// ---------------------------------------------------------------------------
// TDM: async-load a 64-row x 32-col bf16 tile (row stride `ld` elems) from
// global into LDS at byte offset lds_off, padding 2 DWORDs after every 16
// DWORDs (one 64B row) -> LDS row stride 36 bf16 elements (matches LDT).
// Tracked by TENSORcnt. Issued by one wave per workgroup (EXEC is ignored).
// ---------------------------------------------------------------------------
__device__ __forceinline__ void tdm_load_tile64x32(unsigned lds_off,
                                                   const unsigned short* gptr,
                                                   unsigned long long ld_elems)
{
  const unsigned long long ga = (unsigned long long)(uintptr_t)gptr;
  v4u g0;
  g0[0] = 1u;                                        // count=1, user mode
  g0[1] = lds_off;                                   // lds_addr (bytes)
  g0[2] = (unsigned)(ga & 0xFFFFFFFFu);              // global_addr[31:0]
  g0[3] = (unsigned)((ga >> 32) & 0x01FFFFFFu)       // global_addr[56:32]
        | (2u << 30);                                // type = 2 ("image")

  const unsigned d0 = 32u;   // tensor_dim0 == tile_dim0 (in-bounds tile)
  const unsigned d1 = 64u;   // tensor_dim1 == tile_dim1
  v8i g1;
  g1[0] = (int)( (1u << 16)        // data_size = 1 -> 2 bytes
               | (1u << 20)        // pad_enable
               | (3u << 22)        // pad_interval code 3 -> 16 DWORDs
               | (1u << 25) );     // pad_amount  code 1 -> 2 DWORDs
  g1[1] = (int)((d0 & 0xFFFFu) << 16);               // tensor_dim0[15:0] @ bit48
  g1[2] = (int)((d0 >> 16) | ((d1 & 0xFFFFu) << 16));// dim0 hi | dim1 lo
  g1[3] = (int)((d1 >> 16) | (32u << 16));           // dim1 hi | tile_dim0=32
  g1[4] = (int)(64u);                                // tile_dim1=64, tile_dim2=0
  g1[5] = (int)(ld_elems & 0xFFFFFFFFull);           // tensor_dim0_stride lo
  g1[6] = (int)((ld_elems >> 32) & 0xFFFFull);       // stride hi; dim1_stride=0
  g1[7] = 0;

  const v4i gz = {0, 0, 0, 0};
#if defined(__clang_major__) && (__clang_major__ >= 23)
  const v8i gz8 = {0, 0, 0, 0, 0, 0, 0, 0};
  __builtin_amdgcn_tensor_load_to_lds(g0, g1, gz, gz, gz8, 0);
#else
  __builtin_amdgcn_tensor_load_to_lds(g0, g1, gz, gz, 0);
#endif
}

// ---------------------------------------------------------------------------
// Tiled WMMA GEMM (bf16 operands, f32 accumulate):
//   C[m][n] = sum_k A[m][k] * W[n][k]        (C = A @ W^T)
// 128 threads = 4 waves, 64x64 tile, K chunks of 32. Tiles fetched by the
// Tensor Data Mover into double-buffered LDS; wave0 issues, TENSORcnt syncs.
// EPI 0: plain f32 store. EPI 2: x = tanh(acc + bias[n]) * tv[m]; store f32
// and bf16 mirrors (affine/dec_h update).
// ---------------------------------------------------------------------------
static constexpr int TM  = 64;
static constexpr int TN  = 64;
static constexpr int KC  = 32;
static constexpr int LDT = 36;   // LDS row stride in bf16 elems

template<int EPI>
__global__ __launch_bounds__(128)
void wmma_gemm(const unsigned short* __restrict__ A, long lda, long aZ,
               const unsigned short* __restrict__ W, long ldw, long wZ,
               float* __restrict__ C, long ldc, long cZ,
               int K,
               const float* __restrict__ bias,
               const float* __restrict__ tv,
               unsigned short* __restrict__ Cbf)   // bf16 mirror (EPI 2)
{
  __shared__ unsigned short sA[2][TM * LDT];
  __shared__ unsigned short sB[2][TN * LDT];

  const int tid  = threadIdx.x;
  const int lane = tid & 31;
  const int wave = tid >> 5;
  const int wm   = wave >> 1;
  const int wn   = wave & 1;
  const long m0  = (long)blockIdx.y * TM;
  const long n0  = (long)blockIdx.x * TN;
  const long z   = blockIdx.z;

  A += z * aZ;  W += z * wZ;  C += z * cZ;

  const unsigned ldsA0 = (unsigned)(size_t)&sA[0][0];
  const unsigned ldsB0 = (unsigned)(size_t)&sB[0][0];
  const unsigned bufBytes = (unsigned)(TM * LDT * 2);

  v8f acc[4];
  #pragma unroll
  for (int i = 0; i < 4; ++i)
    acc[i] = (v8f){0.f,0.f,0.f,0.f,0.f,0.f,0.f,0.f};

  // prologue: chunk 0 -> buffer 0
  if (wave == 0) {
    tdm_load_tile64x32(ldsA0, A + m0 * lda, (unsigned long long)lda);
    tdm_load_tile64x32(ldsB0, W + n0 * ldw, (unsigned long long)ldw);
    __builtin_amdgcn_s_wait_tensorcnt(0);
  }
  __syncthreads();

  const int nch = K / KC;
  for (int c = 0; c < nch; ++c) {
    const int buf = c & 1;

    // issue next chunk into the other buffer; overlaps with WMMA below
    if (wave == 0 && (c + 1) < nch) {
      const long k1 = (long)(c + 1) * KC;
      tdm_load_tile64x32(ldsA0 + (buf ^ 1) * bufBytes, A + m0 * lda + k1,
                         (unsigned long long)lda);
      tdm_load_tile64x32(ldsB0 + (buf ^ 1) * bufBytes, W + n0 * ldw + k1,
                         (unsigned long long)ldw);
    }

    const unsigned short* pA = &sA[buf][0];
    const unsigned short* pB = &sB[buf][0];

    // ---- fragments: lane half h holds k = 16h..16h+15, same order for A and B ----
    AFragU af[2];
    BFragU bf[2];
    {
      const int kh = (lane < 16) ? 0 : 16;
      #pragma unroll
      for (int fm = 0; fm < 2; ++fm) {
        const int mr = wm * 32 + fm * 16 + (lane & 15);
        #pragma unroll
        for (int j = 0; j < 8; ++j) {
          const int kk = kh + 2 * j;
          af[fm].u[j] = *(const unsigned*)&pA[mr * LDT + kk];
        }
      }
    }
    {
      const int kh = (lane < 16) ? 0 : 16;
      #pragma unroll
      for (int fn = 0; fn < 2; ++fn) {
        const int nr = wn * 32 + fn * 16 + (lane & 15);
        #pragma unroll
        for (int j = 0; j < 8; ++j)
          bf[fn].u[j] = *(const unsigned*)&pB[nr * LDT + kh + 2 * j];
      }
    }

    #pragma unroll
    for (int fm = 0; fm < 2; ++fm)
      #pragma unroll
      for (int fn = 0; fn < 2; ++fn)
        acc[fm * 2 + fn] = __builtin_amdgcn_wmma_f32_16x16x32_bf16(
            false, af[fm].v, false, bf[fn].v,
            (short)0, acc[fm * 2 + fn], false, false);

    if (wave == 0) __builtin_amdgcn_s_wait_tensorcnt(0);
    __syncthreads();
  }

  // ---- epilogue: C/D layout: VGPR v -> M = (lane<16?0:8)+v, N = lane&15 ----
  const int rbase = (lane < 16) ? 0 : 8;
  const int cl    = lane & 15;
  #pragma unroll
  for (int fm = 0; fm < 2; ++fm) {
    #pragma unroll
    for (int fn = 0; fn < 2; ++fn) {
      const long col = n0 + wn * 32 + fn * 16 + cl;
      #pragma unroll
      for (int v = 0; v < 8; ++v) {
        const long row = m0 + wm * 32 + fm * 16 + rbase + v;
        float x = acc[fm * 2 + fn][v];
        if (EPI == 2) {
          x = tanhf(x + bias[col]);
          x *= tv[row];
          Cbf[row * ldc + col] = f2bf(x);
        }
        C[row * ldc + col] = x;
      }
    }
  }
}

// ---------------------------------------------------------------------------
// f32 -> bf16 conversion (weights / feature map mirrors, once per launch)
// ---------------------------------------------------------------------------
__global__ __launch_bounds__(256)
void to_bf16(const float* __restrict__ src, unsigned short* __restrict__ dst,
             long n)
{
  const long i = (long)blockIdx.x * 256 + threadIdx.x;
  if (i < n) dst[i] = f2bf(src[i]);
}

// ---------------------------------------------------------------------------
// GRU gate elementwise kernel:  h' = (1-z)*n + z*h
// Writes f32 (optional) + bf16 hidden, and optional f32+bf16 enc_states slot.
// ---------------------------------------------------------------------------
__global__ __launch_bounds__(256)
void gru_gates(const float* __restrict__ gi, const float* __restrict__ gh,
               const float* __restrict__ bih, const float* __restrict__ bhh,
               const float* __restrict__ hprev, long hprevStride,
               float* __restrict__ hout, unsigned short* __restrict__ hout_bf,
               long houtStride,
               float* __restrict__ es, unsigned short* __restrict__ es_bf,
               long esStride)
{
  const long tid = (long)blockIdx.x * 256 + threadIdx.x;   // < B*H
  const long b = tid >> 8;          // H == 256
  const int  j = (int)(tid & 255);
  const float* giB = gi + b * (3 * H_);
  const float* ghB = gh + b * (3 * H_);
  const float ir  = giB[j]          + bih[j];
  const float iz  = giB[H_ + j]     + bih[H_ + j];
  const float in_ = giB[2 * H_ + j] + bih[2 * H_ + j];
  const float hr  = ghB[j]          + bhh[j];
  const float hz  = ghB[H_ + j]     + bhh[H_ + j];
  const float hn  = ghB[2 * H_ + j] + bhh[2 * H_ + j];
  const float r   = 1.f / (1.f + expf(-(ir + hr)));
  const float zz  = 1.f / (1.f + expf(-(iz + hz)));
  const float n   = tanhf(in_ + r * hn);
  const float h   = (1.f - zz) * n + zz * hprev[b * hprevStride + j];
  if (hout)   hout[b * houtStride + j] = h;
  hout_bf[b * houtStride + j] = f2bf(h);
  if (es)    es[b * esStride + j] = h;
  if (es_bf) es_bf[b * esStride + j] = f2bf(h);
}

// ---------------------------------------------------------------------------
// Fused attention step: one workgroup (256 threads) per batch element.
// ---------------------------------------------------------------------------
__global__ __launch_bounds__(256)
void attention_step(const float* __restrict__ encW1T,     // [B, WD, S]
                    const float* __restrict__ enc_states, // [B, S, H] f32
                    const float* __restrict__ decW2,      // [B, WD]
                    const float* __restrict__ Vv,         // [WD]
                    const float* __restrict__ feature_map,// [B, SEQ, D]
                    const float* __restrict__ term_vec,   // [D]
                    float* __restrict__ mask,             // [B, S]
                    float* __restrict__ tv,               // [B]
                    float* __restrict__ logp,             // [B]
                    unsigned short* __restrict__ cat_bf,  // [B, 2H] bf16
                    unsigned short* __restrict__ dec_in_bf,// [B, D] bf16
                    float* __restrict__ out_sched,        // [B, T] (as float)
                    int t)
{
  const int b = blockIdx.x;
  const int s = threadIdx.x;

  __shared__ float dw[WD_], vsh[WD_], prob[S_], mkd[S_], bv[S_];
  __shared__ int   bi[S_];
  __shared__ float s_tv;
  __shared__ int   s_idx;

  dw[s]  = decW2[(long)b * WD_ + s];
  vsh[s] = Vv[s];
  __syncthreads();

  // scores[s] = sum_w V[w] * tanh(encW1T[b][w][s] + decW2[b][w])  (coalesced)
  const float* ew = encW1T + (long)b * WD_ * S_ + s;
  float acc = 0.f;
  #pragma unroll 4
  for (int w = 0; w < WD_; ++w) {
    if ((w & 31) == 0 && w + 32 < WD_)
      __builtin_prefetch(ew + (long)(w + 32) * S_, 0, 0);
    acc += vsh[w] * tanhf(ew[(long)w * S_] + dw[w]);
  }

  const float m  = mask[(long)b * S_ + s];
  const float sc = (m == 0.f) ? NEG_INF_ : acc;

  // softmax
  bv[s] = sc; __syncthreads();
  for (int off = 128; off; off >>= 1) {
    if (s < off) bv[s] = fmaxf(bv[s], bv[s + off]);
    __syncthreads();
  }
  const float mx = bv[0]; __syncthreads();
  const float e = expf(sc - mx);
  bv[s] = e; __syncthreads();
  for (int off = 128; off; off >>= 1) {
    if (s < off) bv[s] += bv[s + off];
    __syncthreads();
  }
  const float p = e / bv[0];
  __syncthreads();
  prob[s] = p;
  const float md = p * m;
  mkd[s] = md;

  // argmax (first occurrence on ties => strict >)
  bv[s] = md; bi[s] = s; __syncthreads();
  for (int off = 128; off; off >>= 1) {
    if (s < off && bv[s + off] > bv[s]) { bv[s] = bv[s + off]; bi[s] = bi[s + off]; }
    __syncthreads();
  }
  const int idx1 = bi[0]; __syncthreads();

  // second best (exclude idx1)
  bv[s] = (s == idx1) ? -1.f : md; bi[s] = s; __syncthreads();
  for (int off = 128; off; off >>= 1) {
    if (s < off && bv[s + off] > bv[s]) { bv[s] = bv[s + off]; bi[s] = bi[s + off]; }
    __syncthreads();
  }
  const int idx2 = bi[0]; __syncthreads();

  if (s == 0) {
    int idx = idx1;
    if (t < 1 && idx == 0) idx = idx2;              // MIN_T == 1
    const float ip   = mkd[idx];                    // gathered pre-force
    const float m0c  = mask[(long)b * S_ + 0];      // pre-update column 0
    const int   idxf = (m0c == 0.f) ? 0 : idx;
    const float tvn  = tv[b] * m0c;
    mask[(long)b * S_ + idxf] = 0.f;
    tv[b] = tvn;
    logp[b] += tvn * logf(ip + EPS_);
    out_sched[(long)b * T_ + t] = (float)idxf;
    s_tv = tvn; s_idx = idxf;
  }
  __syncthreads();

  // lin[h] = sum_s prob[s] * enc_states[b][s][h]  -> cat_bf[:, H:]
  const float* es = enc_states + (long)b * S_ * H_ + s;
  float lin = 0.f;
  #pragma unroll 4
  for (int ss = 0; ss < S_; ++ss)
    lin += prob[ss] * es[(long)ss * H_];
  cat_bf[(long)b * (2 * H_) + H_ + s] = f2bf(lin);

  // dec_in = aug[b, idxf] * tv_new  (bf16 mirror for next-step GEMM)
  const int   idxf = s_idx;
  const float tvn  = s_tv;
  if (s < D_) {
    const float xv = (idxf == 0)
        ? term_vec[s]
        : feature_map[((long)b * SEQ_ + (idxf - 1)) * D_ + s];
    dec_in_bf[(long)b * D_ + s] = f2bf(xv * tvn);
  }
}

// ---------------------------------------------------------------------------
// Init / finalize
// ---------------------------------------------------------------------------
__global__ __launch_bounds__(256)
void init_state(float* __restrict__ h, unsigned short* __restrict__ h_bf,
                float* __restrict__ mask,
                float* __restrict__ tv, float* __restrict__ logp,
                unsigned short* __restrict__ dec_in_bf,
                const float* __restrict__ dec_input0,
                float* __restrict__ out)
{
  const long tid = (long)blockIdx.x * 256 + threadIdx.x;   // 131072 threads
  if (tid < (long)B_ * H_) { h[tid] = 0.f; h_bf[tid] = 0; }
  if (tid < (long)B_ * S_) { mask[tid] = 1.f; }
  if (tid < (long)B_ * D_) { dec_in_bf[tid] = f2bf(dec_input0[tid & (D_ - 1)]); }
  if (tid < B_) { tv[tid] = 1.f; logp[tid] = 0.f; }
  if (tid == 0) { out[B_ + B_ * T_] = 0.f; }               // trailing scalar 0
}

__global__ __launch_bounds__(256)
void finalize_logp(const float* __restrict__ logp, float* __restrict__ out)
{
  const int b = blockIdx.x * 256 + threadIdx.x;
  if (b < B_) out[b] = logp[b];
}

// ---------------------------------------------------------------------------
// Host-side launch sequence (graph-capture safe: stream launches only)
// ---------------------------------------------------------------------------
extern "C" void kernel_launch(void* const* d_in, const int* in_sizes, int n_in,
                              void* d_out, int out_size, void* d_ws, size_t ws_size,
                              hipStream_t stream)
{
  (void)in_sizes; (void)n_in; (void)out_size; (void)ws_size;

  const float* feature_map = (const float*)d_in[0];
  const float* enc_Wih     = (const float*)d_in[1];
  const float* enc_Whh     = (const float*)d_in[2];
  const float* enc_bih     = (const float*)d_in[3];
  const float* enc_bhh     = (const float*)d_in[4];
  const float* dec_Wih     = (const float*)d_in[5];
  const float* dec_Whh     = (const float*)d_in[6];
  const float* dec_bih     = (const float*)d_in[7];
  const float* dec_bhh     = (const float*)d_in[8];
  const float* W1          = (const float*)d_in[9];
  const float* W2          = (const float*)d_in[10];
  const float* Vv          = (const float*)d_in[11];
  const float* aff_W       = (const float*)d_in[12];
  const float* aff_b       = (const float*)d_in[13];
  const float* dec_input0  = (const float*)d_in[14];
  const float* term_vec    = (const float*)d_in[15];
  float* out = (float*)d_out;

  // ---- workspace layout ----
  float* ws = (float*)d_ws;
  size_t off = 0;
  float* enc_states = ws + off; off += (size_t)B_ * S_ * H_;   // 134 MB
  float* encW1T     = ws + off; off += (size_t)B_ * WD_ * S_;  // 134 MB
  float* h          = ws + off; off += (size_t)B_ * H_;        // dec_h / enc h
  float* gi         = ws + off; off += (size_t)B_ * 3 * H_;
  float* gh         = ws + off; off += (size_t)B_ * 3 * H_;
  float* decW2      = ws + off; off += (size_t)B_ * WD_;
  float* mask       = ws + off; off += (size_t)B_ * S_;
  float* tv         = ws + off; off += B_;
  float* logp       = ws + off; off += B_;
  // bf16 (ushort) region
  unsigned short* us = (unsigned short*)(ws + off);
  size_t uo = 0;
  unsigned short* fm_bf     = us + uo; uo += (size_t)B_ * SEQ_ * D_;  // 33 MB
  unsigned short* term_bf   = us + uo; uo += D_;
  unsigned short* es_bf     = us + uo; uo += (size_t)B_ * S_ * H_;    // 67 MB
  unsigned short* h_bf      = us + uo; uo += (size_t)B_ * H_;
  unsigned short* cat_bf    = us + uo; uo += (size_t)B_ * 2 * H_;
  unsigned short* dec_in_bf = us + uo; uo += (size_t)B_ * D_;
  unsigned short* eWih_bf   = us + uo; uo += (size_t)3 * H_ * D_;
  unsigned short* eWhh_bf   = us + uo; uo += (size_t)3 * H_ * H_;
  unsigned short* dWih_bf   = us + uo; uo += (size_t)3 * H_ * D_;
  unsigned short* dWhh_bf   = us + uo; uo += (size_t)3 * H_ * H_;
  unsigned short* W1_bf     = us + uo; uo += (size_t)WD_ * H_;
  unsigned short* W2_bf     = us + uo; uo += (size_t)WD_ * H_;
  unsigned short* affW_bf   = us + uo; uo += (size_t)H_ * 2 * H_;

  auto cvt = [&](const float* s, unsigned short* d, long n) {
    to_bf16<<<(unsigned)((n + 255) / 256), 256, 0, stream>>>(s, d, n);
  };
  cvt(feature_map, fm_bf, (long)B_ * SEQ_ * D_);
  cvt(term_vec,    term_bf, D_);
  cvt(enc_Wih,     eWih_bf, (long)3 * H_ * D_);
  cvt(enc_Whh,     eWhh_bf, (long)3 * H_ * H_);
  cvt(dec_Wih,     dWih_bf, (long)3 * H_ * D_);
  cvt(dec_Whh,     dWhh_bf, (long)3 * H_ * H_);
  cvt(W1,          W1_bf,   (long)WD_ * H_);
  cvt(W2,          W2_bf,   (long)WD_ * H_);
  cvt(aff_W,       affW_bf, (long)H_ * 2 * H_);

  init_state<<<512, 256, 0, stream>>>(h, h_bf, mask, tv, logp,
                                      dec_in_bf, dec_input0, out);

  const dim3 blk(128);
  const dim3 gruGrid(3 * H_ / TN, B_ / TM, 1);    // 12 x 8

  // ---- encoder recurrence over S=256 steps ----
  for (int t = 0; t < S_; ++t) {
    const unsigned short* A = (t == 0) ? term_bf : fm_bf + (size_t)(t - 1) * D_;
    const long lda = (t == 0) ? 0 : (long)SEQ_ * D_;
    wmma_gemm<0><<<gruGrid, blk, 0, stream>>>(A, lda, 0, eWih_bf, D_, 0,
                                              gi, 3 * H_, 0, D_,
                                              nullptr, nullptr, nullptr);
    wmma_gemm<0><<<gruGrid, blk, 0, stream>>>(h_bf, H_, 0, eWhh_bf, H_, 0,
                                              gh, 3 * H_, 0, H_,
                                              nullptr, nullptr, nullptr);
    gru_gates<<<(B_ * H_) / 256, 256, 0, stream>>>(
        gi, gh, enc_bih, enc_bhh, h, H_,
        h, h_bf, H_,
        enc_states + (size_t)t * H_, es_bf + (size_t)t * H_, (long)S_ * H_);
  }

  // ---- encW1^T[b][w][s] = sum_h W1[w][h] * enc_states[b][s][h] ----
  wmma_gemm<0><<<dim3(S_ / TN, WD_ / TM, B_), blk, 0, stream>>>(
      W1_bf, H_, 0,
      es_bf, H_, (long)S_ * H_,
      encW1T, S_, (long)WD_ * S_,
      H_, nullptr, nullptr, nullptr);

  // ---- decoder over T=32 steps ----
  for (int t = 0; t < T_; ++t) {
    wmma_gemm<0><<<gruGrid, blk, 0, stream>>>(dec_in_bf, D_, 0, dWih_bf, D_, 0,
                                              gi, 3 * H_, 0, D_,
                                              nullptr, nullptr, nullptr);
    wmma_gemm<0><<<gruGrid, blk, 0, stream>>>(h_bf, H_, 0, dWhh_bf, H_, 0,
                                              gh, 3 * H_, 0, H_,
                                              nullptr, nullptr, nullptr);
    // hidden -> cat_bf[:, :H]; dec_h (h/h_bf) left intact (attention uses OLD h)
    gru_gates<<<(B_ * H_) / 256, 256, 0, stream>>>(
        gi, gh, dec_bih, dec_bhh, h, H_,
        nullptr, cat_bf, 2 * H_,
        nullptr, nullptr, 0);
    // decW2 = dec_h @ W2^T
    wmma_gemm<0><<<dim3(WD_ / TN, B_ / TM, 1), blk, 0, stream>>>(
        h_bf, H_, 0, W2_bf, H_, 0, decW2, WD_, 0, H_,
        nullptr, nullptr, nullptr);

    attention_step<<<B_, 256, 0, stream>>>(encW1T, enc_states, decW2, Vv,
                                           feature_map, term_vec,
                                           mask, tv, logp, cat_bf, dec_in_bf,
                                           out + B_, t);

    // dec_h = tanh(cat @ aff_W^T + aff_b) * tv   (fused; writes f32 + bf16)
    wmma_gemm<2><<<dim3(H_ / TN, B_ / TM, 1), blk, 0, stream>>>(
        cat_bf, 2 * H_, 0, affW_bf, 2 * H_, 0, h, H_, 0, 2 * H_,
        aff_b, tv, h_bf);
  }

  finalize_logp<<<2, 256, 0, stream>>>(logp, out);
}
